// RNN_45217415693095
// MI455X (gfx1250) — matrix operand order repaired and checked by reading, not compiled, not run on hardware
//
#include <hip/hip_runtime.h>
#include <hip/hip_bf16.h>
#include <math.h>

// ---------------------------------------------------------------------------
// 2-layer tanh RNN LM for MI455X (gfx1250, wave32, WMMA).
// bf16 operands + fp32 accumulation via v_wmma_f32_16x16x32_bf16.
// Round 2: kill VGPR spills (launch_bounds + 1 wave/SIMD on big GEMMs),
// split-K + LDS-reduction step kernels for the latency-bound recurrence.
// ---------------------------------------------------------------------------

typedef __bf16 bf16;
typedef __attribute__((ext_vector_type(16))) __bf16 v16bf;
typedef __attribute__((ext_vector_type(8)))  __bf16 v8bf;
typedef __attribute__((ext_vector_type(8)))  float  v8f;

#define EMB   1024
#define HID   1024
#define SEQ   128
#define BATCH 64
#define VOCAB 10000
#define VPAD  10240           // VOCAB padded to multiple of 128
#define SB    (SEQ * BATCH)   // 8192

union FragBF { v16bf v; v8bf h[2]; };

// A-fragment (16x32, MxK) from row-major bf16 A.
__device__ __forceinline__ v16bf load_frag_a(const bf16* __restrict__ A, int lda,
                                             int m0, int k0) {
  const int lane = threadIdx.x & 31;
  const int m = lane & 15, hi = lane >> 4;
  const bf16* p = A + (size_t)(m0 + m) * lda + k0 + hi * 8;
  FragBF f;
  f.h[0] = *(const v8bf*)(p);
  f.h[1] = *(const v8bf*)(p + 16);
  return f.v;
}

// B-fragment (32x16, KxN) where B[k][n] = W[n][k], W row-major (N x K).
__device__ __forceinline__ v16bf load_frag_b(const bf16* __restrict__ W, int ldw,
                                             int n0, int k0) {
  const int lane = threadIdx.x & 31;
  const int n = lane & 15, hi = lane >> 4;
  const bf16* p = W + (size_t)(n0 + n) * ldw + k0 + hi * 8;
  FragBF f;
  f.h[0] = *(const v8bf*)(p);
  f.h[1] = *(const v8bf*)(p + 16);
  return f.v;
}

template <int NI, int NJ>
__device__ __forceinline__ void zero_acc(v8f acc[NI][NJ]) {
  const v8f vz = {};
#pragma unroll
  for (int i = 0; i < NI; ++i)
#pragma unroll
    for (int j = 0; j < NJ; ++j) acc[i][j] = vz;
}

// Wave computes a (16*NI)x(16*NJ) tile over K range [k_lo, k_lo+k_len).
template <int NI, int NJ>
__device__ __forceinline__ void gemm_tile(const bf16* __restrict__ A, int lda, int m0,
                                          const bf16* __restrict__ W, int ldw, int n0,
                                          int k_lo, int k_len, v8f acc[NI][NJ]) {
  for (int kk = 0; kk < k_len; kk += 32) {
    const int k0 = k_lo + kk;
    v16bf a[NI], b[NJ];
#pragma unroll
    for (int i = 0; i < NI; ++i) a[i] = load_frag_a(A, lda, m0 + 16 * i, k0);
#pragma unroll
    for (int j = 0; j < NJ; ++j) b[j] = load_frag_b(W, ldw, n0 + 16 * j, k0);
#pragma unroll
    for (int i = 0; i < NI; ++i)
#pragma unroll
      for (int j = 0; j < NJ; ++j)
        acc[i][j] = __builtin_amdgcn_wmma_f32_16x16x32_bf16(
            false, a[i], false, b[j], (short)0, acc[i][j], false, false);
  }
}

// ------------------------------- prep kernels -------------------------------

__global__ void k_f32_to_bf16(const float* __restrict__ src, bf16* __restrict__ dst, int n) {
  int i = blockIdx.x * blockDim.x + threadIdx.x;
  if (i < n) dst[i] = (bf16)src[i];
}

__global__ void k_vw_pad(const float* __restrict__ vw, bf16* __restrict__ dst) {
  int i = blockIdx.x * blockDim.x + threadIdx.x;  // over VPAD*HID
  if (i >= VPAD * HID) return;
  int r = i >> 10;  // / HID
  dst[i] = (r < VOCAB) ? (bf16)vw[i] : (bf16)0.0f;
}

__global__ void k_gather_embed(const int* __restrict__ tok, const float* __restrict__ emb,
                               bf16* __restrict__ out) {
  int i = blockIdx.x * blockDim.x + threadIdx.x;  // over SB*EMB
  if (i >= SB * EMB) return;
  int sb = i >> 10;          // / EMB
  int e  = i & (EMB - 1);
  out[i] = (bf16)emb[(size_t)tok[sb] * EMB + e];
}

// -------------------- X0 = emb @ wx0^T + bh0 (8192x1024) --------------------
// 128 thr = 4 waves (1/SIMD), wave tile 64x64, block tile 128x128.

__global__ __launch_bounds__(128, 1) void
k_x0_gemm(const bf16* __restrict__ embb, const bf16* __restrict__ wx0b,
          const float* __restrict__ bh0, float* __restrict__ X0) {
  const int wave = threadIdx.x >> 5;
  const int m0 = blockIdx.x * 128 + (wave >> 1) * 64;
  const int n0 = blockIdx.y * 128 + (wave & 1) * 64;
  v8f acc[4][4];
  zero_acc<4, 4>(acc);
  gemm_tile<4, 4>(embb, EMB, m0, wx0b, EMB, n0, 0, EMB, acc);

  const int lane = threadIdx.x & 31, n = lane & 15, hi = lane >> 4;
#pragma unroll
  for (int i = 0; i < 4; ++i)
#pragma unroll
    for (int j = 0; j < 4; ++j) {
      const int col = n0 + 16 * j + n;
      const float bias = bh0[col];
#pragma unroll
      for (int r = 0; r < 8; ++r) {
        const int row = m0 + 16 * i + hi * 8 + r;
        X0[(size_t)row * HID + col] = acc[i][j][r] + bias;
      }
    }
}

// ---------------- recurrent steps: split-K + LDS reduction ------------------
// Block: 256 thr = 8 waves. Block owns a 64x32 output tile (n0 = blockIdx.x*32).
// Each wave accumulates a K-slice into acc[4][2]; partials reduced through LDS.

__device__ __forceinline__ void store_partial(float* __restrict__ red, int wave,
                                              const v8f acc[4][2]) {
  const int lane = threadIdx.x & 31, n = lane & 15, hi = lane >> 4;
  float* p = red + wave * (64 * 32);
#pragma unroll
  for (int i = 0; i < 4; ++i)
#pragma unroll
    for (int j = 0; j < 2; ++j) {
      const int col = 16 * j + n;
#pragma unroll
      for (int r = 0; r < 8; ++r)
        p[(16 * i + hi * 8 + r) * 32 + col] = acc[i][j][r];
    }
}

__global__ __launch_bounds__(256, 2) void
k_step0(const bf16* __restrict__ h0_prev, const bf16* __restrict__ wh0b,
        const float* __restrict__ x0s, bf16* __restrict__ h0_cur,
        float* __restrict__ hid_out0) {
  __shared__ float red[8 * 64 * 32];  // 64 KB
  const int wave = threadIdx.x >> 5;
  const int n0 = blockIdx.x * 32;

  v8f acc[4][2];
  zero_acc<4, 2>(acc);
  // wave's K slice: 128 of 1024
  gemm_tile<4, 2>(h0_prev, HID, 0, wh0b, HID, n0, wave * 128, 128, acc);
  store_partial(red, wave, acc);
  __syncthreads();

#pragma unroll
  for (int e = 0; e < 8; ++e) {
    const int idx = threadIdx.x * 8 + e;  // 0..2047 over 64x32 tile
    const int row = idx >> 5;             // batch index
    const int col = (idx & 31) + n0;      // hidden index
    float s = 0.0f;
#pragma unroll
    for (int w = 0; w < 8; ++w) s += red[w * (64 * 32) + idx];
    const float v = tanhf(x0s[row * HID + col] + s);
    h0_cur[row * HID + col]   = (bf16)v;
    hid_out0[row * HID + col] = v;
  }
}

__global__ __launch_bounds__(256, 2) void
k_step1(const bf16* __restrict__ h0_cur, const bf16* __restrict__ h1_prev,
        const bf16* __restrict__ wx1b, const bf16* __restrict__ wh1b,
        const float* __restrict__ bh1, bf16* __restrict__ h1_cur,
        bf16* __restrict__ top_s, float* __restrict__ hid_out1) {
  __shared__ float red[8 * 64 * 32];  // 64 KB
  const int wave = threadIdx.x >> 5;
  const int n0 = blockIdx.x * 32;

  // Concatenated K = [h0 @ wx1^T (1024) | h1_prev @ wh1^T (1024)] -> 8 x 256
  const bf16* A = (wave < 4) ? h0_cur : h1_prev;
  const bf16* W = (wave < 4) ? wx1b   : wh1b;
  const int k_lo = (wave & 3) * 256;

  v8f acc[4][2];
  zero_acc<4, 2>(acc);
  gemm_tile<4, 2>(A, HID, 0, W, HID, n0, k_lo, 256, acc);
  store_partial(red, wave, acc);
  __syncthreads();

#pragma unroll
  for (int e = 0; e < 8; ++e) {
    const int idx = threadIdx.x * 8 + e;
    const int row = idx >> 5;
    const int col = (idx & 31) + n0;
    float s = bh1[col];
#pragma unroll
    for (int w = 0; w < 8; ++w) s += red[w * (64 * 32) + idx];
    const float v = tanhf(s);
    h1_cur[row * HID + col]   = (bf16)v;
    top_s[row * HID + col]    = (bf16)v;
    hid_out1[row * HID + col] = v;
  }
}

// -------------------- logits = top @ v_w^T + v_b (8192x10000) ---------------
// 128 thr = 4 waves, wave tile 64x64, block tile 128x128; NT stores.

__global__ __launch_bounds__(128, 1) void
k_logits(const bf16* __restrict__ topb, const bf16* __restrict__ vwb,
         const float* __restrict__ vb, float* __restrict__ out) {
  const int wave = threadIdx.x >> 5;
  const int m0 = blockIdx.x * 128 + (wave >> 1) * 64;
  const int n0 = blockIdx.y * 128 + (wave & 1) * 64;
  v8f acc[4][4];
  zero_acc<4, 4>(acc);
  gemm_tile<4, 4>(topb, HID, m0, vwb, HID, n0, 0, HID, acc);

  const int lane = threadIdx.x & 31, n = lane & 15, hi = lane >> 4;
#pragma unroll
  for (int i = 0; i < 4; ++i)
#pragma unroll
    for (int j = 0; j < 4; ++j) {
      const int col = n0 + 16 * j + n;
      if (col < VOCAB) {
        const float bias = vb[col];
#pragma unroll
        for (int r = 0; r < 8; ++r) {
          const int row = m0 + 16 * i + hi * 8 + r;
          __builtin_nontemporal_store(acc[i][j][r] + bias,
                                      &out[(size_t)row * VOCAB + col]);
        }
      }
    }
}

// --------------------------------- driver -----------------------------------

extern "C" void kernel_launch(void* const* d_in, const int* in_sizes, int n_in,
                              void* d_out, int out_size, void* d_ws, size_t ws_size,
                              hipStream_t stream) {
  (void)in_sizes; (void)n_in; (void)out_size; (void)ws_size;

  const int*   inputs    = (const int*)  d_in[0];
  const float* hidden    = (const float*)d_in[1];
  const float* embedding = (const float*)d_in[2];
  const float* wx0 = (const float*)d_in[3];
  const float* wh0 = (const float*)d_in[4];
  const float* bh0 = (const float*)d_in[5];
  const float* wx1 = (const float*)d_in[6];
  const float* wh1 = (const float*)d_in[7];
  const float* bh1 = (const float*)d_in[8];
  const float* v_w = (const float*)d_in[9];
  const float* v_b = (const float*)d_in[10];

  float* logits    = (float*)d_out;                       // (S,B,V)
  float* hid_final = logits + (size_t)SB * VOCAB;         // (L,B,H)

  // workspace partition
  char* ws = (char*)d_ws;
  size_t off = 0;
  auto alloc = [&](size_t bytes) -> char* {
    char* p = ws + off;
    off += (bytes + 255) & ~(size_t)255;
    return p;
  };
  bf16*  embb  = (bf16*)alloc((size_t)SB * EMB * sizeof(bf16));
  bf16*  wx0b  = (bf16*)alloc((size_t)HID * EMB * sizeof(bf16));
  bf16*  wh0b  = (bf16*)alloc((size_t)HID * HID * sizeof(bf16));
  bf16*  wx1b  = (bf16*)alloc((size_t)HID * HID * sizeof(bf16));
  bf16*  wh1b  = (bf16*)alloc((size_t)HID * HID * sizeof(bf16));
  bf16*  vwb   = (bf16*)alloc((size_t)VPAD * HID * sizeof(bf16));
  float* X0    = (float*)alloc((size_t)SB * HID * sizeof(float));
  bf16*  topb  = (bf16*)alloc((size_t)SB * HID * sizeof(bf16));
  bf16*  h0buf[2], *h1buf[2];
  h0buf[0] = (bf16*)alloc((size_t)BATCH * HID * sizeof(bf16));
  h0buf[1] = (bf16*)alloc((size_t)BATCH * HID * sizeof(bf16));
  h1buf[0] = (bf16*)alloc((size_t)BATCH * HID * sizeof(bf16));
  h1buf[1] = (bf16*)alloc((size_t)BATCH * HID * sizeof(bf16));

  // ---- prep: weight conversion, vocab pad, embedding gather, hidden init ----
  k_f32_to_bf16<<<(HID * EMB + 255) / 256, 256, 0, stream>>>(wx0, wx0b, HID * EMB);
  k_f32_to_bf16<<<(HID * HID + 255) / 256, 256, 0, stream>>>(wh0, wh0b, HID * HID);
  k_f32_to_bf16<<<(HID * HID + 255) / 256, 256, 0, stream>>>(wx1, wx1b, HID * HID);
  k_f32_to_bf16<<<(HID * HID + 255) / 256, 256, 0, stream>>>(wh1, wh1b, HID * HID);
  k_vw_pad<<<(VPAD * HID + 255) / 256, 256, 0, stream>>>(v_w, vwb);
  k_gather_embed<<<(SB * EMB + 255) / 256, 256, 0, stream>>>(inputs, embedding, embb);
  k_f32_to_bf16<<<(BATCH * HID + 255) / 256, 256, 0, stream>>>(hidden, h0buf[0], BATCH * HID);
  k_f32_to_bf16<<<(BATCH * HID + 255) / 256, 256, 0, stream>>>(hidden + (size_t)BATCH * HID,
                                                               h1buf[0], BATCH * HID);

  // ---- X0 = emb @ wx0^T + bh0 : big parallel GEMM ----
  dim3 gx0(SB / 128, HID / 128);
  k_x0_gemm<<<gx0, 128, 0, stream>>>(embb, wx0b, bh0, X0);

  // ---- sequential scan: 2 kernels per timestep, double-buffered state ----
  for (int s = 0; s < SEQ; ++s) {
    const int p = s & 1;
    k_step0<<<HID / 32, 256, 0, stream>>>(h0buf[p], wh0b, X0 + (size_t)s * BATCH * HID,
                                          h0buf[1 - p], hid_final);
    k_step1<<<HID / 32, 256, 0, stream>>>(h0buf[1 - p], h1buf[p], wx1b, wh1b, bh1,
                                          h1buf[1 - p], topb + (size_t)s * BATCH * HID,
                                          hid_final + (size_t)BATCH * HID);
  }

  // ---- logits = top @ v_w^T + v_b ----
  dim3 glg(SB / 128, VPAD / 128);
  k_logits<<<glg, 128, 0, stream>>>(topb, vwb, v_b, logits);
}
